// ModulUwagi_31748398252617
// MI455X (gfx1250) — matrix operand order repaired and checked by reading, not compiled, not run on hardware
//
#include <hip/hip_runtime.h>

// Problem constants (from reference)
#define B_  8
#define T_  2048
#define C_  1024
#define K_  128

// CDNA5 async global->LDS staging (GLOBAL_LOAD_ASYNC_TO_LDS_B128, ASYNCcnt).
#define USE_ASYNC 1

typedef __attribute__((ext_vector_type(16))) __bf16 v16bf;
typedef __attribute__((ext_vector_type(2)))  __bf16 v2bf;
typedef __attribute__((ext_vector_type(8)))  float  v8f;

union FragB16 { v16bf v; unsigned int u[8]; };

__device__ __forceinline__ unsigned short f2bf(float f) {
    union { __bf16 h; unsigned short s; } cv;
    cv.h = (__bf16)f;                 // native bf16 convert (RNE)
    return cv.s;
}

// packed pair convert: backend matches build_vector(fptrunc,fptrunc) ->
// v_cvt_pk_bf16_f32 on gfx1250
__device__ __forceinline__ unsigned int f2bf2(float lo, float hi) {
    v2bf v;
    v[0] = (__bf16)lo;
    v[1] = (__bf16)hi;
    unsigned int r;
    __builtin_memcpy(&r, &v, 4);
    return r;
}

__device__ __forceinline__ v8f v8f_zero() {
    v8f z;
#pragma unroll
    for (int i = 0; i < 8; ++i) z[i] = 0.0f;
    return z;
}

// butterfly reductions within 16-lane groups (wave32; xor masks 1..8 stay in-group)
__device__ __forceinline__ float grp_max16(float v) {
#pragma unroll
    for (int m = 8; m >= 1; m >>= 1) v = fmaxf(v, __shfl_xor(v, m, 32));
    return v;
}
__device__ __forceinline__ float grp_sum16(float v) {
#pragma unroll
    for (int m = 8; m >= 1; m >>= 1) v += __shfl_xor(v, m, 32);
    return v;
}

// ---- async global->LDS helpers -------------------------------------------
__device__ __forceinline__ void async_ld_b128(unsigned short* lds_dst,
                                              const unsigned short* gsrc) {
#if USE_ASYNC
    unsigned int  dst = (unsigned int)(unsigned long long)lds_dst; // LDS byte offset
    unsigned long long src = (unsigned long long)gsrc;
    asm volatile("global_load_async_to_lds_b128 %0, %1, off"
                 :: "v"(dst), "v"(src) : "memory");
#else
    *(uint4*)lds_dst = *(const uint4*)gsrc;
#endif
}

__device__ __forceinline__ void wait_async_keep4() {
#if USE_ASYNC
#if __has_builtin(__builtin_amdgcn_s_wait_asynccnt)
    __builtin_amdgcn_s_wait_asynccnt(4);
#else
    asm volatile("s_wait_asynccnt 0x4" ::: "memory");
#endif
#endif
}
__device__ __forceinline__ void wait_async_all() {
#if USE_ASYNC
#if __has_builtin(__builtin_amdgcn_s_wait_asynccnt)
    __builtin_amdgcn_s_wait_asynccnt(0);
#else
    asm volatile("s_wait_asynccnt 0x0" ::: "memory");
#endif
#endif
}

// ---------------------------------------------------------------------------
// Kernel 1: q/k/v projection.  grid = (M/128, 3), block = 256 (8 waves).
//   widx 0 -> q row-major bf16 [B*T][128]
//   widx 1 -> k row-major bf16 [B*T][128]
//   widx 2 -> v transposed bf16, per batch [128][T]
// ---------------------------------------------------------------------------
__global__ __launch_bounds__(256)
void qkv_proj_kernel(const float* __restrict__ x,
                     const float* __restrict__ Wq,
                     const float* __restrict__ Wk,
                     const float* __restrict__ Wv,
                     unsigned short* __restrict__ ws) {
    __shared__ __align__(16) unsigned short lds_x[128 * 32];  // [row][kk] bf16
    __shared__ __align__(16) unsigned short lds_w[128 * 32];  // [n][kk]  bf16 (transposed)

    const int widx    = blockIdx.y;
    const float* W    = (widx == 0) ? Wq : (widx == 1) ? Wk : Wv;
    const int rowbase = blockIdx.x * 128;
    const int tid     = threadIdx.x;
    const int wave    = __builtin_amdgcn_readfirstlane(tid >> 5);
    const int lane    = tid & 31;
    const int half    = lane >> 4;
    const int l16     = lane & 15;

    v8f acc[8];
#pragma unroll
    for (int n = 0; n < 8; ++n) acc[n] = v8f_zero();

    for (int k0 = 0; k0 < C_; k0 += 32) {
        // prefetch next x tile into near caches (WGP scope)
        if (k0 + 32 < C_) {
            const float* nx = &x[(size_t)(rowbase + (tid >> 1)) * C_ + k0 + 32 + (tid & 1) * 16];
            __builtin_prefetch(nx, 0, 3);
        }
        // x tile [128][32] fp32 -> bf16 LDS, pairwise (float2 load, packed cvt,
        // single ds_store_b32 per pair)
#pragma unroll
        for (int e = 0; e < 2048; e += 256) {
            int p  = e + tid;                 // pair index
            int r  = p >> 4;
            int c2 = (p & 15) << 1;
            float2 f = *(const float2*)&x[(size_t)(rowbase + r) * C_ + k0 + c2];
            *(unsigned int*)&lds_x[r * 32 + c2] = f2bf2(f.x, f.y);
        }
        // W tile [32][128] fp32 -> transposed bf16 LDS [n][kk]; pick the two
        // k-neighbors of each slot so the pair is contiguous in LDS
#pragma unroll
        for (int e = 0; e < 2048; e += 256) {
            int p  = e + tid;
            int n  = p & 127;
            int kk = (p >> 7) << 1;
            float lo = W[(size_t)(k0 + kk)     * K_ + n];
            float hi = W[(size_t)(k0 + kk + 1) * K_ + n];
            *(unsigned int*)&lds_w[n * 32 + kk] = f2bf2(lo, hi);
        }
        __syncthreads();

        // A fragment: 16x32 bf16; VGPR j holds K = 16*(j>>2) + 8*half + 2*(j&3)
        FragB16 a;
        const int arow = wave * 16 + l16;
#pragma unroll
        for (int j = 0; j < 8; ++j) {
            int kb = ((j >> 2) << 4) + (half << 3) + ((j & 3) << 1);
            a.u[j] = *(const unsigned int*)&lds_x[arow * 32 + kb];
        }

#pragma unroll
        for (int n = 0; n < 8; ++n) {
            // B fragment: 32x16 bf16; VGPR j holds K = 16*(j>>2) + 8*half + 2*(j&3), N = l16
            FragB16 b;
            const int col = n * 16 + l16;
#pragma unroll
            for (int j = 0; j < 8; ++j) {
                int kb = ((j >> 2) << 4) + (half << 3) + ((j & 3) << 1);
                b.u[j] = *(const unsigned int*)&lds_w[col * 32 + kb];
            }
            acc[n] = __builtin_amdgcn_wmma_f32_16x16x32_bf16(
                false, a.v, false, b.v, (short)0, acc[n], false, false);
        }
        __syncthreads();
    }

    // store: C/D layout -> row = i + 8*half, col = n*16 + l16
    if (widx < 2) {
        unsigned short* dst = ws + (size_t)widx * ((size_t)B_ * T_ * K_);
#pragma unroll
        for (int n = 0; n < 8; ++n)
#pragma unroll
            for (int i = 0; i < 8; ++i) {
                int row = rowbase + wave * 16 + i + (half << 3);
                int col = n * 16 + l16;
                dst[(size_t)row * K_ + col] = f2bf(acc[n][i]);
            }
    } else {
        unsigned short* vt = ws + (size_t)2 * ((size_t)B_ * T_ * K_);
#pragma unroll
        for (int n = 0; n < 8; ++n)
#pragma unroll
            for (int i = 0; i < 8; ++i) {
                int row = rowbase + wave * 16 + i + (half << 3);  // global token
                int col = n * 16 + l16;                           // head dim
                int b   = row >> 11;          // / T_
                int t   = row & (T_ - 1);
                vt[(size_t)b * (K_ * T_) + (size_t)col * T_ + t] = f2bf(acc[n][i]);
            }
    }
}

// ---------------------------------------------------------------------------
// Kernel 2: causal flash attention.  grid = (T/128, B), block = 256 (8 waves).
// Block-uniform key loop; K/V tiles staged cooperatively into LDS with
// double-buffered async copies; each wave owns 16 query rows.
// ---------------------------------------------------------------------------
__device__ __forceinline__ void stage_kv(unsigned short* lk, unsigned short* lv,
                                         const unsigned short* kb,
                                         const unsigned short* vb,
                                         int kt, int tid) {
    // K tile: [32 keys][128 hd] bf16 = 8KB, contiguous in kb
#pragma unroll
    for (int r = 0; r < 2; ++r) {
        int chunk = tid + r * 256;                       // 512 chunks x 16B
        async_ld_b128(lk + chunk * 8, kb + (size_t)kt * K_ + (size_t)chunk * 8);
    }
    // V tile from vT: [128 hd][32 keys] bf16 = 8KB (row segments of 64B)
#pragma unroll
    for (int r = 0; r < 2; ++r) {
        int e  = tid + r * 256;
        int hd = e >> 2, seg = e & 3;
        async_ld_b128(lv + hd * 32 + seg * 8,
                      vb + (size_t)hd * T_ + kt + seg * 8);
    }
}

__global__ __launch_bounds__(256)
void attn_kernel(const unsigned short* __restrict__ ws,
                 float* __restrict__ out) {
    __shared__ __align__(16) unsigned short lds_k[2][32 * 128];
    __shared__ __align__(16) unsigned short lds_v[2][128 * 32];
    __shared__ __align__(16) unsigned short lds_p[8][16 * 32];

    const int qtile = blockIdx.x;
    const int bat   = blockIdx.y;
    const int tid   = threadIdx.x;
    const int wave  = __builtin_amdgcn_readfirstlane(tid >> 5);
    const int lane  = tid & 31;
    const int half  = lane >> 4;
    const int l16   = lane & 15;
    const int qrow0   = qtile * 128 + wave * 16;   // first query row (in batch)
    const int qrow_hi = qrow0 + 16;                // causal key bound (exclusive)

    const unsigned short* qb = ws + (size_t)bat * T_ * K_;
    const unsigned short* kb = ws + (size_t)(B_ * T_ * K_) + (size_t)bat * T_ * K_;
    const unsigned short* vb = ws + (size_t)2 * (B_ * T_ * K_) + (size_t)bat * K_ * T_;

    // Q fragments: 4 chunks of 16x32 over head dim 128
    FragB16 qf[4];
#pragma unroll
    for (int c = 0; c < 4; ++c)
#pragma unroll
        for (int j = 0; j < 8; ++j) {
            int kq = c * 32 + ((j >> 2) << 4) + (half << 3) + ((j & 3) << 1);
            qf[c].u[j] = *(const unsigned int*)&qb[(size_t)(qrow0 + l16) * K_ + kq];
        }

    v8f acc[8];
#pragma unroll
    for (int n = 0; n < 8; ++n) acc[n] = v8f_zero();
    float m_i[8], l_i[8];
#pragma unroll
    for (int i = 0; i < 8; ++i) { m_i[i] = -1e30f; l_i[i] = 0.0f; }

    const float scale = 0.08838834764831845f;  // 1/sqrt(128)

    const int ntiles = qtile * 4 + 4;          // block-uniform causal bound
    stage_kv(lds_k[0], lds_v[0], kb, vb, 0, tid);

    for (int it = 0; it < ntiles; ++it) {
        const int kt  = it * 32;
        const int cur = it & 1;
        const bool more = (it + 1) < ntiles;
        if (more) stage_kv(lds_k[cur ^ 1], lds_v[cur ^ 1], kb, vb, kt + 32, tid);
        if (more) wait_async_keep4(); else wait_async_all();
        __syncthreads();   // tile `cur` fully in LDS for all waves

        if (kt < qrow_hi) {  // wave-uniform causal skip
            const unsigned short* lk = lds_k[cur];
            const unsigned short* lv = lds_v[cur];

            // ---- S = Q * K^T for two 16-key subtiles ----
            v8f s0 = v8f_zero(), s1 = v8f_zero();
#pragma unroll
            for (int c = 0; c < 4; ++c) {
                FragB16 b0, b1;
#pragma unroll
                for (int j = 0; j < 8; ++j) {
                    int hd = c * 32 + ((j >> 2) << 4) + (half << 3) + ((j & 3) << 1);
                    b0.u[j] = *(const unsigned int*)&lk[l16 * K_ + hd];
                    b1.u[j] = *(const unsigned int*)&lk[(16 + l16) * K_ + hd];
                }
                s0 = __builtin_amdgcn_wmma_f32_16x16x32_bf16(
                    false, qf[c].v, false, b0.v, (short)0, s0, false, false);
                s1 = __builtin_amdgcn_wmma_f32_16x16x32_bf16(
                    false, qf[c].v, false, b1.v, (short)0, s1, false, false);
            }

            // ---- scale + causal mask ----
            const int key0 = kt + l16;
            const int key1 = kt + 16 + l16;
#pragma unroll
            for (int i = 0; i < 8; ++i) {
                int row = qrow0 + i + (half << 3);
                s0[i] = (key0 <= row) ? s0[i] * scale : -1e30f;
                s1[i] = (key1 <= row) ? s1[i] * scale : -1e30f;
            }

            // ---- online softmax per row slot ----
            float alpha[8];
#pragma unroll
            for (int i = 0; i < 8; ++i) {
                float tmax = grp_max16(fmaxf(s0[i], s1[i]));
                float mnew = fmaxf(m_i[i], tmax);
                alpha[i] = __expf(m_i[i] - mnew);
                m_i[i] = mnew;
                float p0 = __expf(s0[i] - mnew);
                float p1 = __expf(s1[i] - mnew);
                l_i[i] = l_i[i] * alpha[i] + grp_sum16(p0 + p1);
                lds_p[wave][(i + (half << 3)) * 32 + l16]      = f2bf(p0);
                lds_p[wave][(i + (half << 3)) * 32 + 16 + l16] = f2bf(p1);
            }

            // rescale running output
#pragma unroll
            for (int n = 0; n < 8; ++n)
#pragma unroll
                for (int i = 0; i < 8; ++i) acc[n][i] = acc[n][i] * alpha[i];

            // ---- P fragment (A-layout) from per-wave LDS ----
            FragB16 pf;
#pragma unroll
            for (int j = 0; j < 8; ++j) {
                int kp = ((j >> 2) << 4) + (half << 3) + ((j & 3) << 1);
                pf.u[j] = *(const unsigned int*)&lds_p[wave][l16 * 32 + kp];
            }

            // ---- O += P * V ----
#pragma unroll
            for (int n = 0; n < 8; ++n) {
                FragB16 vf;
                const int hd = n * 16 + l16;
#pragma unroll
                for (int j = 0; j < 8; ++j) {
                    int klocal = ((j >> 2) << 4) + (half << 3) + ((j & 3) << 1);
                    vf.u[j] = *(const unsigned int*)&lv[hd * 32 + klocal];
                }
                acc[n] = __builtin_amdgcn_wmma_f32_16x16x32_bf16(
                    false, pf.v, false, vf.v, (short)0, acc[n], false, false);
            }
        }
        __syncthreads();   // all waves done with tile `cur` before overwrite
    }

    // ---- normalize + store fp32 output ----
#pragma unroll
    for (int i = 0; i < 8; ++i) {
        float inv = 1.0f / l_i[i];
        int row = qrow0 + i + (half << 3);
#pragma unroll
        for (int n = 0; n < 8; ++n) {
            int col = n * 16 + l16;
            out[((size_t)bat * T_ + row) * K_ + col] = acc[n][i] * inv;
        }
    }
}

// ---------------------------------------------------------------------------
extern "C" void kernel_launch(void* const* d_in, const int* in_sizes, int n_in,
                              void* d_out, int out_size, void* d_ws, size_t ws_size,
                              hipStream_t stream) {
    const float* x  = (const float*)d_in[0];
    const float* Wq = (const float*)d_in[1];
    const float* Wk = (const float*)d_in[2];
    const float* Wv = (const float*)d_in[3];
    unsigned short* ws = (unsigned short*)d_ws;   // q | k | vT, bf16, 12 MB total
    float* out = (float*)d_out;

    dim3 g1((B_ * T_) / 128, 3, 1);
    qkv_proj_kernel<<<g1, 256, 0, stream>>>(x, Wq, Wk, Wv, ws);

    dim3 g2(T_ / 128, B_, 1);
    attn_kernel<<<g2, 256, 0, stream>>>(ws, out);
}
